// SE3TransformerWrapper_38963943309435
// MI455X (gfx1250) — compile-verified, matrix operand order checked by reading, not run on hardware
//
#include <hip/hip_runtime.h>
#include <hip/hip_bf16.h>
#include <math.h>

typedef __attribute__((ext_vector_type(16))) __bf16 v16bf;
typedef __attribute__((ext_vector_type(8)))  __bf16 v8bf;
typedef __attribute__((ext_vector_type(8)))  float  v8f;

constexpr int kD   = 256;
constexpr int kH   = 4;
constexpr int kDH  = 64;
constexpr int kL   = 2;
constexpr int kNB  = 10;
constexpr int kED  = 51;
constexpr int kEIN = 70;   // kED + kNB + 9
constexpr int kKE  = 96;   // padded K for edge-feature GEMM (multiple of 32)
constexpr int kIN  = 512;

// GEMM tiling
constexpr int kTW  = 8;    // 16x16 WMMA tiles per wave (wave tile 16 x 128)
constexpr int kBM  = 128;  // block tile M (8 waves x 16 rows)
constexpr int kBN  = 128;  // block tile N
constexpr int kBK  = 32;   // K step per WMMA
constexpr int kLS  = 40;   // padded LDS row stride (bf16 elems) -> bank-conflict free

// ---------------------------------------------------------------------------
// small helpers
// ---------------------------------------------------------------------------
__device__ __forceinline__ float atomicMaxFloat(float* addr, float val) {
  int* ia = (int*)addr;
  int old = __float_as_int(*addr);
  while (__int_as_float(old) < val) {
    int assumed = old;
    old = atomicCAS(ia, assumed, __float_as_int(val));
    if (old == assumed) break;
  }
  return __int_as_float(old);
}

__device__ __forceinline__ float silu_f(float x) {
  return x / (1.0f + __expf(-x));
}

__device__ __forceinline__ v16bf combine16(v8bf lo, v8bf hi) {
  v16bf o;
#pragma unroll
  for (int i = 0; i < 8; ++i) { o[i] = lo[i]; o[i + 8] = hi[i]; }
  return o;
}

// ---------------------------------------------------------------------------
// Generic bf16 WMMA GEMM:
//   C[M,Ncols] = act(A[M,K] @ BT[Ncols,K]^T + bias) (+ Cin residual)
// Block: 256 thr / 8 waves. Block tile 128(M) x 128(N); wave tile 16 x 128.
// B K-slab staged via LDS (double-buffered across barriers); A fragment
// ping/pong in registers (K loop unrolled x2 -> no register copies, no
// WMMA->VALU hazard NOPs); B fragment software-pipelined one tile ahead so
// every ds_load_b128 overlaps the previous v_wmma.
// ---------------------------------------------------------------------------
__global__ __launch_bounds__(256)
void wmma_gemm_kernel(const __bf16* __restrict__ A, int lda,
                      const __bf16* __restrict__ BT, int ldb,
                      const float* __restrict__ bias,
                      const float* __restrict__ Cin,
                      float* __restrict__ Cout,
                      __bf16* __restrict__ Cbout,
                      int M, int Ncols, int K, int act) {
  __shared__ __attribute__((aligned(16))) __bf16 Bs[2][kBN * kLS];

  const int lane = threadIdx.x & 31;
  const int w    = threadIdx.x >> 5;
  const int row0 = blockIdx.x * kBM + w * 16;
  const int col0 = blockIdx.y * kBN;
  const int sel  = lane >> 4;          // ISA 16-bit A/B layout K-half select

  // per-lane A row (clamped so tail blocks stay in-bounds; stores are guarded)
  const int rA = min(row0 + (lane & 15), M - 1);
  const __bf16* aRow = A + (size_t)rA * lda + sel * 8;

  const int nk = K / kBK;
  const int t  = threadIdx.x;

  // cooperative stage of B slab [kBN rows x kBK K] into LDS buffer `buf`
  auto stage = [&](int kt, int buf) {
#pragma unroll
    for (int c = 0; c < 2; ++c) {
      int chunk = t + c * 256;               // 512 x 16B chunks
      int r     = chunk >> 2;
      int koff  = (chunk & 3) * 8;
      v8bf d = *(const v8bf*)(BT + (size_t)(col0 + r) * ldb + kt * kBK + koff);
      *(v8bf*)(&Bs[buf][r * kLS + koff]) = d;
    }
  };

  auto loadA = [&](int kt) {
    const __bf16* p = aRow + kt * kBK;
    return combine16(*(const v8bf*)p, *(const v8bf*)(p + 16));
  };

  auto loadB = [&](int buf, int nt) {
    const __bf16* bp = &Bs[buf][(nt * 16 + (lane & 15)) * kLS + sel * 8];
    return combine16(*(const v8bf*)bp, *(const v8bf*)(bp + 16));
  };

  v8f zero = {};
  v8f acc[kTW];
#pragma unroll
  for (int i = 0; i < kTW; ++i) acc[i] = zero;

  // one K-step: 8 WMMAs with B fragment pipelined one tile ahead
  auto compute = [&](int buf, v16bf aC) {
    v16bf bC = loadB(buf, 0);
#pragma unroll
    for (int nt = 0; nt < kTW; ++nt) {
      v16bf bN = bC;
      if (nt + 1 < kTW) bN = loadB(buf, nt + 1);
      acc[nt] = __builtin_amdgcn_wmma_f32_16x16x32_bf16(
          /*neg_a=*/false, aC, /*neg_b=*/false, bC,
          /*c_mod=*/(short)0, acc[nt], /*reuse_a=*/false, /*reuse_b=*/false);
      bC = bN;
    }
  };

  stage(0, 0);
  v16bf aPing = loadA(0);
  v16bf aPong = aPing;
  __syncthreads();

  int kt = 0;
#pragma unroll 1
  for (; kt + 2 <= nk; kt += 2) {
    // even step: LDS buf 0, A = aPing
    if (kt + 1 < nk) { stage(kt + 1, 1); aPong = loadA(kt + 1); }
    if (kt + 2 < nk) __builtin_prefetch(aRow + (kt + 2) * kBK, 0, 3);
    compute(0, aPing);
    __syncthreads();
    // odd step: LDS buf 1, A = aPong
    if (kt + 2 < nk) { stage(kt + 2, 0); aPing = loadA(kt + 2); }
    if (kt + 3 < nk) __builtin_prefetch(aRow + (kt + 3) * kBK, 0, 3);
    compute(1, aPong);
    __syncthreads();
  }
  if (nk & 1) {
    // remaining step: kt == nk-1 (even), LDS buf 0, A = aPing
    compute(0, aPing);
  }

  // epilogue: C/D layout -> VGPR r holds row (row0 + r + 8*sel), col lane&15
#pragma unroll
  for (int nt = 0; nt < kTW; ++nt) {
    int col = col0 + nt * 16 + (lane & 15);
    float bv = bias ? bias[col] : 0.0f;
#pragma unroll
    for (int r = 0; r < 8; ++r) {
      int row = row0 + r + sel * 8;
      if (row < M) {
        size_t o = (size_t)row * Ncols + col;
        float vv = acc[nt][r] + bv;
        if (act == 1) vv = silu_f(vv);
        if (Cin)   vv += Cin[o];
        if (Cout)  Cout[o]  = vv;
        if (Cbout) Cbout[o] = (__bf16)vv;
      }
    }
  }
}

// ---------------------------------------------------------------------------
// elementwise utilities
// ---------------------------------------------------------------------------
__global__ void cvt_f32_bf16_kernel(const float* __restrict__ in,
                                    __bf16* __restrict__ out, size_t n) {
  size_t i = (size_t)blockIdx.x * blockDim.x + threadIdx.x;
  if (i < n) out[i] = (__bf16)in[i];
}

__global__ void copy_f32_kernel(const float* __restrict__ in,
                                float* __restrict__ out, size_t n) {
  size_t i = (size_t)blockIdx.x * blockDim.x + threadIdx.x;
  if (i < n) out[i] = in[i];
}

__global__ void fill_f32_kernel(float* __restrict__ p, float v, size_t n) {
  size_t i = (size_t)blockIdx.x * blockDim.x + threadIdx.x;
  if (i < n) p[i] = v;
}

// W [K, Ncols] row-major (f32)  ->  WT [Ncols, Kp] row-major (bf16, zero pad)
__global__ void transpose_cvt_kernel(const float* __restrict__ W,
                                     __bf16* __restrict__ WT,
                                     int K, int Ncols, int Kp) {
  int idx = blockIdx.x * blockDim.x + threadIdx.x;
  if (idx >= Ncols * Kp) return;
  int n = idx / Kp, kk = idx % Kp;
  WT[idx] = (__bf16)((kk < K) ? W[(size_t)kk * Ncols + n] : 0.0f);
}

__global__ void deg_kernel(const int* __restrict__ dst, float* __restrict__ deg,
                           int E) {
  int e = blockIdx.x * blockDim.x + threadIdx.x;
  if (e < E) atomicAdd(&deg[dst[e]], 1.0f);
}

// ---------------------------------------------------------------------------
// per-edge geometric features -> bf16 rows [E, kKE] (padded), plus rel / 1/(d+1)
// ---------------------------------------------------------------------------
__global__ void edge_feat_kernel(const float* __restrict__ posw,
                                 const float* __restrict__ edge_attr,
                                 const int* __restrict__ src,
                                 const int* __restrict__ dst,
                                 __bf16* __restrict__ efb,
                                 float* __restrict__ rel,
                                 float* __restrict__ dsc, int E) {
  int e = blockIdx.x * blockDim.x + threadIdx.x;
  if (e >= E) return;
  int s = src[e], d = dst[e];
  float rx = posw[d * 3 + 0] - posw[s * 3 + 0];
  float ry = posw[d * 3 + 1] - posw[s * 3 + 1];
  float rz = posw[d * 3 + 2] - posw[s * 3 + 2];
  float dd = sqrtf(rx * rx + ry * ry + rz * rz + 1e-8f);
  rel[e * 3 + 0] = rx; rel[e * 3 + 1] = ry; rel[e * 3 + 2] = rz;
  dsc[e] = 1.0f / (dd + 1.0f);
  float inv = 1.0f / dd;
  float ux = rx * inv, uy = ry * inv, uz = rz * inv;

  __bf16* row = efb + (size_t)e * kKE;
  const float* ea = edge_attr + (size_t)e * kED;
  int j = 0;
#pragma unroll 1
  for (int i = 0; i < kED; ++i) row[j++] = (__bf16)ea[i];
#pragma unroll
  for (int i = 0; i < kNB; ++i) {
    float c = (10.0f / 9.0f) * (float)i;
    float tt = dd - c;
    row[j++] = (__bf16)__expf(-tt * tt);
  }
  row[j++] = (__bf16)1.0f;
  row[j++] = (__bf16)ux;
  row[j++] = (__bf16)uy;
  row[j++] = (__bf16)uz;
  row[j++] = (__bf16)(ux * uy);
  row[j++] = (__bf16)(uy * uz);
  row[j++] = (__bf16)(3.0f * uz * uz - 1.0f);
  row[j++] = (__bf16)(ux * uz);
  row[j++] = (__bf16)(ux * ux - uy * uy);
  for (; j < kKE; ++j) row[j] = (__bf16)0.0f;
}

// ---------------------------------------------------------------------------
// wave-per-edge attention logits: qk dot per head + e @ Wb, atomicMax into m
// lane owns 8 contiguous channels; 8 lanes == one head (DH=64)
// ---------------------------------------------------------------------------
__global__ __launch_bounds__(256)
void logits_kernel(const float* __restrict__ q, const float* __restrict__ k,
                   const __bf16* __restrict__ eb, const float* __restrict__ Wb,
                   const int* __restrict__ src, const int* __restrict__ dst,
                   float* __restrict__ logits, float* __restrict__ m, int E) {
  int wave = (blockIdx.x * blockDim.x + threadIdx.x) >> 5;
  if (wave >= E) return;
  int lane = threadIdx.x & 31;
  int s = src[wave], d = dst[wave];
  const float*  qp = q  + (size_t)d * kD + lane * 8;
  const float*  kp = k  + (size_t)s * kD + lane * 8;
  const __bf16* ep = eb + (size_t)wave * kD + lane * 8;

  float qk = 0.0f;
  float ewb0 = 0.0f, ewb1 = 0.0f, ewb2 = 0.0f, ewb3 = 0.0f;
#pragma unroll
  for (int j = 0; j < 8; ++j) {
    int c = lane * 8 + j;
    qk += qp[j] * kp[j];
    float ev = (float)ep[j];
    ewb0 += ev * Wb[c * kH + 0];
    ewb1 += ev * Wb[c * kH + 1];
    ewb2 += ev * Wb[c * kH + 2];
    ewb3 += ev * Wb[c * kH + 3];
  }
#pragma unroll
  for (int o = 1; o < 8; o <<= 1) qk += __shfl_xor(qk, o, 32);
#pragma unroll
  for (int o = 1; o < 32; o <<= 1) {
    ewb0 += __shfl_xor(ewb0, o, 32);
    ewb1 += __shfl_xor(ewb1, o, 32);
    ewb2 += __shfl_xor(ewb2, o, 32);
    ewb3 += __shfl_xor(ewb3, o, 32);
  }
  if ((lane & 7) == 0) {
    int h = lane >> 3;
    float ewb = (h == 0) ? ewb0 : (h == 1) ? ewb1 : (h == 2) ? ewb2 : ewb3;
    float lg = qk * 0.125f + ewb;  // 1/sqrt(64)
    logits[(size_t)wave * kH + h] = lg;
    atomicMaxFloat(&m[(size_t)d * kH + h], lg);
  }
}

// thread per (edge, head): a = exp(logit - m[dst]); z[dst] += a
__global__ void expsum_kernel(float* __restrict__ logits,
                              const float* __restrict__ m,
                              const int* __restrict__ dst,
                              float* __restrict__ z, int EH) {
  int i = blockIdx.x * blockDim.x + threadIdx.x;
  if (i >= EH) return;
  int e = i >> 2, h = i & 3;
  float a = __expf(logits[i] - m[(size_t)dst[e] * kH + h]);
  logits[i] = a;
  atomicAdd(&z[(size_t)dst[e] * kH + h], a);
}

// wave-per-edge: scatter attn * v[src] * e into agg; fused tanh(e@Wc) pos delta
__global__ __launch_bounds__(256)
void message_kernel(const float* __restrict__ v, const __bf16* __restrict__ eb,
                    const float* __restrict__ a, const float* __restrict__ z,
                    const float* __restrict__ Wc,
                    const int* __restrict__ src, const int* __restrict__ dst,
                    const float* __restrict__ rel, const float* __restrict__ dsc,
                    float* __restrict__ agg, float* __restrict__ dpos, int E) {
  int wave = (blockIdx.x * blockDim.x + threadIdx.x) >> 5;
  if (wave >= E) return;
  int lane = threadIdx.x & 31;
  int s = src[wave], d = dst[wave];
  int h = lane >> 3;
  float attn = a[(size_t)wave * kH + h] / (z[(size_t)d * kH + h] + 1e-9f);

  const float*  vp = v  + (size_t)s * kD + lane * 8;
  const __bf16* ep = eb + (size_t)wave * kD + lane * 8;
  float* ap = agg + (size_t)d * kD + lane * 8;

  float ewc = 0.0f;
#pragma unroll
  for (int j = 0; j < 8; ++j) {
    int c = lane * 8 + j;
    float ev = (float)ep[j];
    atomicAdd(&ap[j], attn * vp[j] * ev);
    ewc += ev * Wc[c];
  }
#pragma unroll
  for (int o = 1; o < 32; o <<= 1) ewc += __shfl_xor(ewc, o, 32);
  if (lane == 0) {
    float w = tanhf(ewc) * dsc[wave];
    atomicAdd(&dpos[(size_t)d * 3 + 0], w * rel[(size_t)wave * 3 + 0]);
    atomicAdd(&dpos[(size_t)d * 3 + 1], w * rel[(size_t)wave * 3 + 1]);
    atomicAdd(&dpos[(size_t)d * 3 + 2], w * rel[(size_t)wave * 3 + 2]);
  }
}

__global__ void pos_update_kernel(float* __restrict__ posw,
                                  const float* __restrict__ dpos,
                                  const float* __restrict__ deg, int N) {
  int n = blockIdx.x * blockDim.x + threadIdx.x;
  if (n >= N) return;
  float inv = 1.0f / (deg[n] + 1.0f);
#pragma unroll
  for (int i = 0; i < 3; ++i) posw[n * 3 + i] += dpos[n * 3 + i] * inv;
}

// ---------------------------------------------------------------------------
// host-side orchestration
// ---------------------------------------------------------------------------
extern "C" void kernel_launch(void* const* d_in, const int* in_sizes, int n_in,
                              void* d_out, int out_size, void* d_ws, size_t ws_size,
                              hipStream_t stream) {
  const float* h_in      = (const float*)d_in[0];
  const float* edge_attr = (const float*)d_in[1];
  const float* pos_in    = (const float*)d_in[2];
  const float* Wproj     = (const float*)d_in[3];
  const float* bproj     = (const float*)d_in[4];
  const float* Wq        = (const float*)d_in[5];
  const float* Wk        = (const float*)d_in[6];
  const float* Wv        = (const float*)d_in[7];
  const float* Wo        = (const float*)d_in[8];
  const float* We        = (const float*)d_in[9];
  const float* be        = (const float*)d_in[10];
  const float* Wb        = (const float*)d_in[11];
  const float* Wc        = (const float*)d_in[12];
  const float* W1        = (const float*)d_in[13];
  const float* b1        = (const float*)d_in[14];
  const float* W2        = (const float*)d_in[15];
  const float* b2        = (const float*)d_in[16];
  const int*   edge_index = (const int*)d_in[17];

  const int N = in_sizes[0] / kIN;
  const int E = in_sizes[17] / 2;
  const int* src = edge_index;
  const int* dst = edge_index + E;

  // ---- workspace bump allocator -------------------------------------------
  char* ws = (char*)d_ws;
  size_t off = 0;
  auto alloc = [&](size_t bytes) -> void* {
    void* p = ws + off;
    off = (off + bytes + 255) & ~(size_t)255;
    return p;
  };
  float*  x     = (float*) alloc((size_t)N * kD * 4);
  __bf16* xb    = (__bf16*)alloc((size_t)N * kD * 2);
  __bf16* hb    = (__bf16*)alloc((size_t)N * kIN * 2);
  float*  q     = (float*) alloc((size_t)N * kD * 4);
  float*  k     = (float*) alloc((size_t)N * kD * 4);
  float*  v     = (float*) alloc((size_t)N * kD * 4);
  float*  agg   = (float*) alloc((size_t)N * kD * 4);
  __bf16* aggb  = (__bf16*)alloc((size_t)N * kD * 2);
  __bf16* tb    = (__bf16*)alloc((size_t)N * kD * 2);
  __bf16* eb    = (__bf16*)alloc((size_t)E * kD * 2);
  __bf16* efb   = (__bf16*)alloc((size_t)E * kKE * 2);
  float*  rel   = (float*) alloc((size_t)E * 3 * 4);
  float*  dsc   = (float*) alloc((size_t)E * 4);
  float*  lgts  = (float*) alloc((size_t)E * kH * 4);
  float*  m     = (float*) alloc((size_t)N * kH * 4);
  float*  z     = (float*) alloc((size_t)N * kH * 4);
  float*  deg   = (float*) alloc((size_t)N * 4);
  float*  posw  = (float*) alloc((size_t)N * 3 * 4);
  float*  dpos  = (float*) alloc((size_t)N * 3 * 4);
  __bf16* WprojT = (__bf16*)alloc((size_t)kD * kIN * 2);
  __bf16* WqT[kL], *WkT[kL], *WvT[kL], *WoT[kL], *W1T[kL], *W2T[kL], *WeT[kL];
  for (int l = 0; l < kL; ++l) {
    WqT[l] = (__bf16*)alloc((size_t)kD * kD * 2);
    WkT[l] = (__bf16*)alloc((size_t)kD * kD * 2);
    WvT[l] = (__bf16*)alloc((size_t)kD * kD * 2);
    WoT[l] = (__bf16*)alloc((size_t)kD * kD * 2);
    W1T[l] = (__bf16*)alloc((size_t)kD * kD * 2);
    W2T[l] = (__bf16*)alloc((size_t)kD * kD * 2);
    WeT[l] = (__bf16*)alloc((size_t)kD * kKE * 2);
  }

  const int TB = 256;
  auto blocks = [](size_t n, int tb) { return (unsigned)((n + tb - 1) / tb); };

  // ---- one-time prep -------------------------------------------------------
  cvt_f32_bf16_kernel<<<blocks((size_t)N * kIN, TB), TB, 0, stream>>>(h_in, hb, (size_t)N * kIN);
  copy_f32_kernel<<<blocks((size_t)N * 3, TB), TB, 0, stream>>>(pos_in, posw, (size_t)N * 3);
  fill_f32_kernel<<<blocks((size_t)N, TB), TB, 0, stream>>>(deg, 0.0f, (size_t)N);
  deg_kernel<<<blocks((size_t)E, TB), TB, 0, stream>>>(dst, deg, E);

  transpose_cvt_kernel<<<blocks((size_t)kD * kIN, TB), TB, 0, stream>>>(Wproj, WprojT, kIN, kD, kIN);
  for (int l = 0; l < kL; ++l) {
    const int dd = kD * kD;
    transpose_cvt_kernel<<<blocks((size_t)kD * kD, TB), TB, 0, stream>>>(Wq + l * dd, WqT[l], kD, kD, kD);
    transpose_cvt_kernel<<<blocks((size_t)kD * kD, TB), TB, 0, stream>>>(Wk + l * dd, WkT[l], kD, kD, kD);
    transpose_cvt_kernel<<<blocks((size_t)kD * kD, TB), TB, 0, stream>>>(Wv + l * dd, WvT[l], kD, kD, kD);
    transpose_cvt_kernel<<<blocks((size_t)kD * kD, TB), TB, 0, stream>>>(Wo + l * dd, WoT[l], kD, kD, kD);
    transpose_cvt_kernel<<<blocks((size_t)kD * kD, TB), TB, 0, stream>>>(W1 + l * dd, W1T[l], kD, kD, kD);
    transpose_cvt_kernel<<<blocks((size_t)kD * kD, TB), TB, 0, stream>>>(W2 + l * dd, W2T[l], kD, kD, kD);
    transpose_cvt_kernel<<<blocks((size_t)kD * kKE, TB), TB, 0, stream>>>(We + l * kEIN * kD, WeT[l], kEIN, kD, kKE);
  }

  auto gemm = [&](const __bf16* A, int lda, const __bf16* BT, int ldb,
                  const float* bias, const float* Cin, float* Cout,
                  __bf16* Cbout, int M, int K, int act) {
    dim3 grid((M + kBM - 1) / kBM, kD / kBN);
    wmma_gemm_kernel<<<grid, 256, 0, stream>>>(A, lda, BT, ldb, bias, Cin,
                                               Cout, Cbout, M, kD, K, act);
  };

  // x = h @ Wproj + bproj   (also emit bf16 copy)
  gemm(hb, kIN, WprojT, kIN, bproj, nullptr, x, xb, N, kIN, 0);

  for (int l = 0; l < kL; ++l) {
    // edge features (geometry from current posw)
    edge_feat_kernel<<<blocks((size_t)E, TB), TB, 0, stream>>>(
        posw, edge_attr, src, dst, efb, rel, dsc, E);
    // e = silu(ef @ We + be) -> bf16 [E, D] (164 MB, fits global L2)
    gemm(efb, kKE, WeT[l], kKE, be + l * kD, nullptr, nullptr, eb, E, kKE, 1);
    // q, k, v
    gemm(xb, kD, WqT[l], kD, nullptr, nullptr, q, nullptr, N, kD, 0);
    gemm(xb, kD, WkT[l], kD, nullptr, nullptr, k, nullptr, N, kD, 0);
    gemm(xb, kD, WvT[l], kD, nullptr, nullptr, v, nullptr, N, kD, 0);
    // reset segment accumulators
    fill_f32_kernel<<<blocks((size_t)N * kH, TB), TB, 0, stream>>>(m, -3.0e38f, (size_t)N * kH);
    fill_f32_kernel<<<blocks((size_t)N * kH, TB), TB, 0, stream>>>(z, 0.0f, (size_t)N * kH);
    fill_f32_kernel<<<blocks((size_t)N * kD, TB), TB, 0, stream>>>(agg, 0.0f, (size_t)N * kD);
    fill_f32_kernel<<<blocks((size_t)N * 3, TB), TB, 0, stream>>>(dpos, 0.0f, (size_t)N * 3);
    // attention
    logits_kernel<<<blocks((size_t)E * 32, 256), 256, 0, stream>>>(
        q, k, eb, Wb + l * kD * kH, src, dst, lgts, m, E);
    expsum_kernel<<<blocks((size_t)E * kH, TB), TB, 0, stream>>>(lgts, m, dst, z, E * kH);
    message_kernel<<<blocks((size_t)E * 32, 256), 256, 0, stream>>>(
        v, eb, lgts, z, Wc + l * kD, src, dst, rel, dsc, agg, dpos, E);
    // x = x + agg @ Wo
    cvt_f32_bf16_kernel<<<blocks((size_t)N * kD, TB), TB, 0, stream>>>(agg, aggb, (size_t)N * kD);
    gemm(aggb, kD, WoT[l], kD, nullptr, x, x, xb, N, kD, 0);
    // FFN: x = x + silu(x @ W1 + b1) @ W2 + b2
    gemm(xb, kD, W1T[l], kD, b1 + l * kD, nullptr, nullptr, tb, N, kD, 1);
    gemm(tb, kD, W2T[l], kD, b2 + l * kD, x, x, xb, N, kD, 0);
    // position update
    pos_update_kernel<<<blocks((size_t)N, TB), TB, 0, stream>>>(posw, dpos, deg, N);
  }

  // outputs: concat(x [N*D], pos [N*3])
  float* out = (float*)d_out;
  copy_f32_kernel<<<blocks((size_t)N * kD, TB), TB, 0, stream>>>(x, out, (size_t)N * kD);
  copy_f32_kernel<<<blocks((size_t)N * 3, TB), TB, 0, stream>>>(posw, out + (size_t)N * kD, (size_t)N * 3);
}
